// FoldBlock_33535104647863
// MI455X (gfx1250) — matrix operand order repaired and checked by reading, not compiled, not run on hardware
//
#include <hip/hip_runtime.h>

// ---------------- CDNA5 WMMA plumbing (wave32, 16x16x32 bf16) ----------------
typedef __attribute__((ext_vector_type(16))) __bf16 v16bf;
typedef __attribute__((ext_vector_type(8)))  float  v8f;

union Frag {
  v16bf v;
  unsigned short u[16];
  uint4 q[2];
};

// Native converts: clang lowers to v_cvt_pk_bf16_f32 on gfx1250.
__device__ __forceinline__ unsigned short f2bf(float f) {
  union { __bf16 b; unsigned short u; } x;
  x.b = (__bf16)f;
  return x.u;
}
__device__ __forceinline__ unsigned int f2bf_pk(float lo, float hi) {
  union { __bf16 b[2]; unsigned int u; } x;
  x.b[0] = (__bf16)lo; x.b[1] = (__bf16)hi;
  return x.u;
}

__device__ __forceinline__ v8f wmma_bf16(const Frag& a, const Frag& b, v8f c) {
  // emits v_wmma_f32_16x16x32_bf16
  return __builtin_amdgcn_wmma_f32_16x16x32_bf16(false, a.v, false, b.v,
                                                 (short)0, c, false, false);
}

// Problem constants
#define NRES 768
#define NHEAD 12
#define PROJW 1152     // 192 + 384 + 144 + 432
#define NFIN 2112
#define NOUT2 384

// ---------------------------------------------------------------------------
// Generic row-major GEMM  C[m, coff+n] = sum_k A[m,k]*B[k,n] + bias[n]
// fp32 in, fp32 out, bf16 WMMA inner.  64x64 tile, 128 threads (4 waves).
// Block-uniform fast path for fully in-bounds tiles (no per-element guards).
// ---------------------------------------------------------------------------
__global__ void __launch_bounds__(128)
gemm_wmma_kernel(const float* __restrict__ A, const float* __restrict__ Bw,
                 const float* __restrict__ bias, float* __restrict__ C,
                 int M, int N, int K, int lda, int ldb, int ldc, int coff)
{
  __shared__ __attribute__((aligned(16))) unsigned short As[64][40];
  __shared__ __attribute__((aligned(16))) unsigned short Bt[64][40]; // col-major
  const int tid  = threadIdx.x;
  const int wave = tid >> 5, lane = tid & 31, half = lane >> 4, r = lane & 15;
  const int m0 = blockIdx.x * 64, n0 = blockIdx.y * 64;
  const bool full = (m0 + 64 <= M) && (n0 + 64 <= N);
  v8f acc[4] = {};

  for (int k0 = 0; k0 < K; k0 += 32) {
    if (full) {
      // A tile: adjacent-k pair -> one packed convert + 32-bit LDS store
      for (int idx = tid * 2; idx < 2048; idx += 256) {
        int row = idx >> 5, k = idx & 31;
        const float* ap = &A[(size_t)(m0 + row) * lda + k0 + k];
        if (k0 + 32 < K) __builtin_prefetch(ap + 32, 0, 1);
        *(unsigned int*)&As[row][k] = f2bf_pk(ap[0], ap[1]);
      }
      // B tile transposed: pack (k, k+1) pair per column into one 32-bit store
      for (int idx = tid; idx < 1024; idx += 128) {
        int k = (idx >> 6) * 2, nn = idx & 63;
        const float* bp = &Bw[(size_t)(k0 + k) * ldb + n0 + nn];
        *(unsigned int*)&Bt[nn][k] = f2bf_pk(bp[0], bp[ldb]);
      }
    } else {
      for (int idx = tid * 2; idx < 2048; idx += 256) {
        int row = idx >> 5, k = idx & 31;
        const float* ap = &A[(size_t)(m0 + row) * lda + k0 + k];
        float a0 = (m0 + row < M) ? ap[0] : 0.f;
        float a1 = (m0 + row < M) ? ap[1] : 0.f;
        *(unsigned int*)&As[row][k] = f2bf_pk(a0, a1);
      }
      for (int idx = tid; idx < 2048; idx += 128) {
        int k = idx >> 6, nn = idx & 63;
        float b = (n0 + nn < N) ? Bw[(size_t)(k0 + k) * ldb + n0 + nn] : 0.f;
        Bt[nn][k] = f2bf(b);
      }
    }
    __syncthreads();
    Frag fa;
    fa.q[0] = *(const uint4*)&As[wave * 16 + r][half * 8];
    fa.q[1] = *(const uint4*)&As[wave * 16 + r][16 + half * 8];
#pragma unroll
    for (int s = 0; s < 4; ++s) {
      Frag fb;
      fb.q[0] = *(const uint4*)&Bt[s * 16 + r][half * 8];
      fb.q[1] = *(const uint4*)&Bt[s * 16 + r][16 + half * 8];
      acc[s] = wmma_bf16(fa, fb, acc[s]);
    }
    __syncthreads();
  }
#pragma unroll
  for (int s = 0; s < 4; ++s)
#pragma unroll
    for (int j = 0; j < 8; ++j) {
      int row = m0 + wave * 16 + half * 8 + j;
      int col = n0 + s * 16 + r;
      if (row < M && col < N)
        C[(size_t)row * ldc + coff + col] = acc[s][j] + bias[col];
    }
}

// ---------------------------------------------------------------------------
// Rotation helper
// ---------------------------------------------------------------------------
__device__ __forceinline__ void quat_rot(const float* quat, int n, float* Rm) {
  float qw = quat[n * 4 + 0], qx = quat[n * 4 + 1];
  float qy = quat[n * 4 + 2], qz = quat[n * 4 + 3];
  float inv = rsqrtf(qw * qw + qx * qx + qy * qy + qz * qz);
  qw *= inv; qx *= inv; qy *= inv; qz *= inv;
  Rm[0] = 1.f - 2.f * (qy * qy + qz * qz); Rm[1] = 2.f * (qx * qy - qw * qz); Rm[2] = 2.f * (qx * qz + qw * qy);
  Rm[3] = 2.f * (qx * qy + qw * qz); Rm[4] = 1.f - 2.f * (qx * qx + qz * qz); Rm[5] = 2.f * (qy * qz - qw * qx);
  Rm[6] = 2.f * (qx * qz - qw * qy); Rm[7] = 2.f * (qy * qz + qw * qx); Rm[8] = 1.f - 2.f * (qx * qx + qy * qy);
}

// ---------------------------------------------------------------------------
// Prep: build bf16 qfeat/kfeat (32 wide: 16 scalar + 12 point + 4 pad),
// vcat (48 wide: 16 v_s + 24 v_pt + 8 pad), and qq/kk softmax biases.
// One thread per (head, residue).
// ---------------------------------------------------------------------------
__global__ void prep_kernel(const float* __restrict__ P, const float* __restrict__ quat,
                            const float* __restrict__ trans, const float* __restrict__ pw_raw,
                            unsigned short* __restrict__ qfeat, unsigned short* __restrict__ kfeat,
                            unsigned short* __restrict__ vcat, float* __restrict__ qq2,
                            float* __restrict__ kk2)
{
  int g = blockIdx.x * 256 + threadIdx.x;
  if (g >= NRES * NHEAD) return;
  int h = g / NRES, n = g % NRES;
  float Rm[9]; quat_rot(quat, n, Rm);
  float tv[3] = { trans[n * 3], trans[n * 3 + 1], trans[n * 3 + 2] };
  float pw = sqrtf(1.f / 54.f) * log1pf(__expf(pw_raw[h]));   // (1/(3*PQK*4.5))^.5 * softplus
  const float sw = 0.14433756729740643f;                      // (1/(3*SQK))^.5
  const float* Pr = P + (size_t)n * PROJW;
  size_t fo = (size_t)h * NRES + n;
  unsigned short* qf = qfeat + fo * 32;
  unsigned short* kf = kfeat + fo * 32;
  unsigned short* vc = vcat + fo * 48;
  for (int c = 0; c < 16; ++c) {
    qf[c] = f2bf(Pr[h * 16 + c] * sw);              // q_s * scalar_w
    kf[c] = f2bf(Pr[192 + h * 32 + c]);             // k_s
    vc[c] = f2bf(Pr[192 + h * 32 + 16 + c]);        // v_s
  }
  float qq = 0.f, kk = 0.f;
  for (int i = 0; i < 3; ++i) {
    for (int p = 0; p < 4; ++p) {                   // q points -> global frame
      float a = tv[i];
      for (int j = 0; j < 3; ++j) a += Rm[i * 3 + j] * Pr[576 + j * 48 + h * 4 + p];
      qf[16 + i * 4 + p] = f2bf(a * pw);            // scaled by point_w for cross term
      qq += a * a;
    }
    for (int c = 0; c < 12; ++c) {                  // kv points -> global frame
      float a = tv[i];
      for (int j = 0; j < 3; ++j) a += Rm[i * 3 + j] * Pr[720 + j * 144 + h * 12 + c];
      if (c < 4) { kf[16 + i * 4 + c] = f2bf(a); kk += a * a; }
      else       { vc[16 + i * 8 + (c - 4)] = f2bf(a); }
    }
  }
  for (int c = 28; c < 32; ++c) { qf[c] = 0; kf[c] = 0; }
  for (int c = 40; c < 48; ++c) vc[c] = 0;
  qq2[h * NRES + n] = 0.5f * pw * qq;
  kk2[h * NRES + n] = 0.5f * pw * kk;
}

// ---------------------------------------------------------------------------
// a2d[row=(n*768+m)][h] = (inputs_2d[row,:] @ w2d[:,h] + b2d[h]) / sqrt(3)
// WMMA GEMM: M=589824 rows, K=128, N=16 (12 used).  8 waves x 16 rows.
// ---------------------------------------------------------------------------
__global__ void __launch_bounds__(256)
a2d_kernel(const float* __restrict__ in2d, const float* __restrict__ w2d,
           const float* __restrict__ b2d, float* __restrict__ a2dbuf)
{
  __shared__ __attribute__((aligned(16))) unsigned short w2t[16][136]; // [h][k]
  int tid = threadIdx.x, wave = tid >> 5, lane = tid & 31, half = lane >> 4, r = lane & 15;
  for (int idx = tid; idx < 2048; idx += 256) {
    int k = idx >> 4, hc = idx & 15;
    float v = (hc < 12) ? w2d[k * 12 + hc] : 0.f;
    w2t[hc][k] = f2bf(v);
  }
  __syncthreads();
  Frag fb[4];
#pragma unroll
  for (int ks = 0; ks < 4; ++ks) {
    fb[ks].q[0] = *(const uint4*)&w2t[r][ks * 32 + half * 8];
    fb[ks].q[1] = *(const uint4*)&w2t[r][ks * 32 + 16 + half * 8];
  }
  size_t rowbase = (size_t)blockIdx.x * 128 + wave * 16;
  const float* p = in2d + (rowbase + r) * 128;
  __builtin_prefetch(p, 0, 1);
  v8f acc = {};
#pragma unroll
  for (int ks = 0; ks < 4; ++ks) {
    Frag fa;
#pragma unroll
    for (int i = 0; i < 4; ++i) {
      *(unsigned int*)&fa.u[2 * i] =
          f2bf_pk(p[ks * 32 + half * 8 + 2 * i], p[ks * 32 + half * 8 + 2 * i + 1]);
      *(unsigned int*)&fa.u[8 + 2 * i] =
          f2bf_pk(p[ks * 32 + 16 + half * 8 + 2 * i], p[ks * 32 + 16 + half * 8 + 2 * i + 1]);
    }
    acc = wmma_bf16(fa, fb[ks], acc);
  }
  int h = r;
  if (h < 12) {
    float bb = b2d[h];
#pragma unroll
    for (int j = 0; j < 8; ++j) {
      size_t row = rowbase + half * 8 + j;
      a2dbuf[row * 16 + h] = (acc[j] + bb) * 0.5773502691896258f;
    }
  }
}

// ---------------------------------------------------------------------------
// Fused logits + softmax for one (head, 16-query tile).
// logits = WMMA(qfeat,kfeat^T) - qq - kk + a2d - HUGE*(1-mask2d); softmax -> bf16 attn
// ---------------------------------------------------------------------------
__global__ void __launch_bounds__(256)
softmax_kernel(const unsigned short* __restrict__ qfeat,
               const unsigned short* __restrict__ kfeat,
               const float* __restrict__ qq2, const float* __restrict__ kk2,
               const float* __restrict__ a2dbuf, const float* __restrict__ mask,
               unsigned short* __restrict__ attn)
{
  __shared__ float S[16][776];
  __shared__ float red[16][16];
  __shared__ float invs[16];
  int tid = threadIdx.x, wave = tid >> 5, lane = tid & 31, half = lane >> 4, r = lane & 15;
  int h = blockIdx.y, n0 = blockIdx.x * 16;

  Frag fq;
  {
    const unsigned short* qp = qfeat + ((size_t)h * NRES + n0 + r) * 32;
    fq.q[0] = *(const uint4*)(qp + half * 8);
    fq.q[1] = *(const uint4*)(qp + 16 + half * 8);
  }
  for (int mb = 0; mb < 6; ++mb) {
    int m0 = mb * 128 + wave * 16;
    Frag fk;
    const unsigned short* kp = kfeat + ((size_t)h * NRES + m0 + r) * 32;
    fk.q[0] = *(const uint4*)(kp + half * 8);
    fk.q[1] = *(const uint4*)(kp + 16 + half * 8);
    v8f acc = {};
    acc = wmma_bf16(fq, fk, acc);
    int m = m0 + r;
    float kkv = kk2[h * NRES + m];
    float mm = mask[m];
#pragma unroll
    for (int j = 0; j < 8; ++j) {
      int nrow = n0 + half * 8 + j;
      float qqv = qq2[h * NRES + nrow];
      float av  = a2dbuf[((size_t)nrow * NRES + m) * 16 + h];
      float mn  = mask[nrow];
      S[half * 8 + j][m] = acc[j] + av - qqv - kkv - 65504.f * (1.f - mn * mm);
    }
  }
  __syncthreads();
  int row = tid >> 4, sub = tid & 15;
  float mx = -3.0e38f;
  for (int c = sub; c < NRES; c += 16) mx = fmaxf(mx, S[row][c]);
  red[row][sub] = mx;
  __syncthreads();
#pragma unroll
  for (int i = 0; i < 16; ++i) mx = fmaxf(mx, red[row][i]);
  __syncthreads();
  float sum = 0.f;
  for (int c = sub; c < NRES; c += 16) {
    float e = __expf(S[row][c] - mx);
    S[row][c] = e; sum += e;
  }
  red[row][sub] = sum;
  __syncthreads();
  if (sub == 0) {
    float tot = 0.f;
#pragma unroll
    for (int i = 0; i < 16; ++i) tot += red[row][i];
    invs[row] = 1.f / tot;
  }
  __syncthreads();
  // packed bf16 writeback: 2 probs per 32-bit store (v_cvt_pk_bf16_f32)
  for (int rw = 0; rw < 16; ++rw) {
    float inv = invs[rw];
    unsigned short* ap = attn + ((size_t)h * NRES + n0 + rw) * NRES;
    for (int c2 = tid; c2 < NRES / 2; c2 += 256)
      *(unsigned int*)(ap + c2 * 2) = f2bf_pk(S[rw][c2 * 2] * inv, S[rw][c2 * 2 + 1] * inv);
  }
}

// ---------------------------------------------------------------------------
// res_s (16) and res_pt_global (24) = attn[h] (768x768) @ vcat[h] (768x48)
// 4 waves x 64 query rows per block, 3 N-subtiles, K loop 24x32.
// ---------------------------------------------------------------------------
__global__ void __launch_bounds__(128)
resv_kernel(const unsigned short* __restrict__ attn, const unsigned short* __restrict__ vcat,
            float* __restrict__ finalbuf, float* __restrict__ resptg)
{
  __shared__ __attribute__((aligned(16))) unsigned short Vt[48][40]; // [col][k]
  int tid = threadIdx.x, wave = tid >> 5, lane = tid & 31, half = lane >> 4, r = lane & 15;
  int h = blockIdx.y, n0 = blockIdx.x * 64;
  v8f acc[3] = {};
  for (int k0 = 0; k0 < NRES; k0 += 32) {
    for (int idx = tid; idx < 1536; idx += 128) {
      int kk = idx / 48, c = idx % 48;
      Vt[c][kk] = vcat[((size_t)h * NRES + k0 + kk) * 48 + c];
    }
    __syncthreads();
    Frag fa;
    const unsigned short* ap = attn + ((size_t)h * NRES + n0 + wave * 16 + r) * NRES + k0;
    fa.q[0] = *(const uint4*)(ap + half * 8);
    fa.q[1] = *(const uint4*)(ap + 16 + half * 8);
#pragma unroll
    for (int s = 0; s < 3; ++s) {
      Frag fb;
      fb.q[0] = *(const uint4*)&Vt[s * 16 + r][half * 8];
      fb.q[1] = *(const uint4*)&Vt[s * 16 + r][16 + half * 8];
      acc[s] = wmma_bf16(fa, fb, acc[s]);
    }
    __syncthreads();
  }
#pragma unroll
  for (int s = 0; s < 3; ++s)
#pragma unroll
    for (int j = 0; j < 8; ++j) {
      int n = n0 + wave * 16 + half * 8 + j;
      int col = s * 16 + r;
      if (col < 16)      finalbuf[(size_t)n * NFIN + h * 16 + col] = acc[s][j];
      else if (col < 40) resptg[((size_t)h * NRES + n) * 24 + col - 16] = acc[s][j];
    }
}

// ---------------------------------------------------------------------------
// res_2d[n] = attn[:, n, :] (16x768) @ inputs_2d[n] (768x128); one block per n.
// attn head rows 12..15 are zero (memset).  8 waves = 8 column tiles of 16.
// ---------------------------------------------------------------------------
__global__ void __launch_bounds__(256)
res2d_kernel(const unsigned short* __restrict__ attn, const float* __restrict__ in2d,
             float* __restrict__ finalbuf)
{
  __shared__ __attribute__((aligned(16))) unsigned short T[128][40]; // [c][k]
  int tid = threadIdx.x, wave = tid >> 5, lane = tid & 31, half = lane >> 4, r = lane & 15;
  int n = blockIdx.x;
  v8f acc = {};
  for (int k0 = 0; k0 < NRES; k0 += 32) {
    // stage 32x128 fp32 -> bf16, packing adjacent k pairs into 32-bit LDS stores
    for (int idx = tid; idx < 2048; idx += 256) {
      int kk = (idx >> 7) * 2, c = idx & 127;
      const float* s0 = &in2d[((size_t)n * NRES + k0 + kk) * 128 + c];
      if (k0 + 32 < NRES) __builtin_prefetch(s0 + 32 * 128, 0, 1);
      *(unsigned int*)&T[c][kk] = f2bf_pk(s0[0], s0[128]);
    }
    __syncthreads();
    Frag fa;  // rows = heads
    const unsigned short* ap = attn + ((size_t)r * NRES + n) * NRES + k0;
    fa.q[0] = *(const uint4*)(ap + half * 8);
    fa.q[1] = *(const uint4*)(ap + 16 + half * 8);
    Frag fb;
    fb.q[0] = *(const uint4*)&T[wave * 16 + r][half * 8];
    fb.q[1] = *(const uint4*)&T[wave * 16 + r][16 + half * 8];
    acc = wmma_bf16(fa, fb, acc);
    __syncthreads();
  }
#pragma unroll
  for (int j = 0; j < 8; ++j) {
    int hh = half * 8 + j;
    if (hh < 12)
      finalbuf[(size_t)n * NFIN + 576 + hh * 128 + wave * 16 + r] = acc[j];
  }
}

// ---------------------------------------------------------------------------
// Inverse-frame transform of point results + norms. One thread per (h, n).
// ---------------------------------------------------------------------------
__global__ void ptlocal_kernel(const float* __restrict__ resptg, const float* __restrict__ quat,
                               const float* __restrict__ trans, float* __restrict__ finalbuf)
{
  int g = blockIdx.x * 256 + threadIdx.x;
  if (g >= NRES * NHEAD) return;
  int h = g / NRES, n = g % NRES;
  float Rm[9]; quat_rot(quat, n, Rm);
  float t0 = trans[n * 3], t1 = trans[n * 3 + 1], t2 = trans[n * 3 + 2];
  const float* src = resptg + ((size_t)h * NRES + n) * 24;
  float* fb = finalbuf + (size_t)n * NFIN;
  for (int p = 0; p < 8; ++p) {
    float gx = src[p] - t0, gy = src[8 + p] - t1, gz = src[16 + p] - t2;
    float l0 = Rm[0] * gx + Rm[3] * gy + Rm[6] * gz;   // R^T
    float l1 = Rm[1] * gx + Rm[4] * gy + Rm[7] * gz;
    float l2 = Rm[2] * gx + Rm[5] * gy + Rm[8] * gz;
    fb[192 + 0 * 96 + h * 8 + p] = l0;
    fb[192 + 1 * 96 + h * 8 + p] = l1;
    fb[192 + 2 * 96 + h * 8 + p] = l2;
    fb[480 + h * 8 + p] = sqrtf(1e-8f + l0 * l0 + l1 * l1 + l2 * l2);
  }
}

// ---------------------------------------------------------------------------
extern "C" void kernel_launch(void* const* d_in, const int* in_sizes, int n_in,
                              void* d_out, int out_size, void* d_ws, size_t ws_size,
                              hipStream_t stream) {
  const float* in1d   = (const float*)d_in[0];
  const float* in2d   = (const float*)d_in[1];
  const float* mask   = (const float*)d_in[2];
  const float* quat   = (const float*)d_in[3];
  const float* trans  = (const float*)d_in[4];
  const float* wq     = (const float*)d_in[5];
  const float* bq     = (const float*)d_in[6];
  const float* wkv    = (const float*)d_in[7];
  const float* bkv    = (const float*)d_in[8];
  const float* wqp    = (const float*)d_in[9];
  const float* bqp    = (const float*)d_in[10];
  const float* wkvp   = (const float*)d_in[11];
  const float* bkvp   = (const float*)d_in[12];
  const float* pw_raw = (const float*)d_in[13];
  const float* w2d    = (const float*)d_in[14];
  const float* b2d    = (const float*)d_in[15];
  const float* wout   = (const float*)d_in[16];
  const float* bout   = (const float*)d_in[17];
  float* out = (float*)d_out;

  char* ws = (char*)d_ws;
  size_t off = 0;
  auto salloc = [&](size_t bytes) -> void* {
    void* p = ws + off;
    off = (off + bytes + 255) & ~(size_t)255;
    return p;
  };
  float*          P        = (float*)salloc((size_t)NRES * PROJW * 4);      // 3.5 MB
  unsigned short* qfeat    = (unsigned short*)salloc((size_t)NHEAD * NRES * 32 * 2);
  unsigned short* kfeat    = (unsigned short*)salloc((size_t)NHEAD * NRES * 32 * 2);
  unsigned short* vcat     = (unsigned short*)salloc((size_t)NHEAD * NRES * 48 * 2);
  float*          qq2      = (float*)salloc((size_t)NHEAD * NRES * 4);
  float*          kk2      = (float*)salloc((size_t)NHEAD * NRES * 4);
  float*          a2dbuf   = (float*)salloc((size_t)NRES * NRES * 16 * 4);  // 37.7 MB
  unsigned short* attn     = (unsigned short*)salloc((size_t)16 * NRES * NRES * 2); // 18.9 MB
  float*          resptg   = (float*)salloc((size_t)NHEAD * NRES * 24 * 4);
  float*          finalbuf = (float*)salloc((size_t)NRES * NFIN * 4);       // 6.5 MB
  (void)ws_size; (void)in_sizes; (void)n_in; (void)out_size;

  // 1) Projections: P[768][1152] = inputs_1d @ [wq|wkv|wqp|wkvp] + biases
  gemm_wmma_kernel<<<dim3(12, 3), 128, 0, stream>>>(in1d, wq,   bq,   P, NRES, 192, 384, 384, 192, PROJW, 0);
  gemm_wmma_kernel<<<dim3(12, 6), 128, 0, stream>>>(in1d, wkv,  bkv,  P, NRES, 384, 384, 384, 384, PROJW, 192);
  gemm_wmma_kernel<<<dim3(12, 3), 128, 0, stream>>>(in1d, wqp,  bqp,  P, NRES, 144, 384, 384, 144, PROJW, 576);
  gemm_wmma_kernel<<<dim3(12, 7), 128, 0, stream>>>(in1d, wkvp, bkvp, P, NRES, 432, 384, 384, 432, PROJW, 720);

  // 2) Frame transforms + bf16 feature packing
  prep_kernel<<<36, 256, 0, stream>>>(P, quat, trans, pw_raw, qfeat, kfeat, vcat, qq2, kk2);

  // 3) Pair bias: first streaming pass over inputs_2d
  a2d_kernel<<<4608, 256, 0, stream>>>(in2d, w2d, b2d, a2dbuf);

  // 4) Zero attn (head pad rows 12..15 stay zero), then fused logits+softmax
  hipMemsetAsync(attn, 0, (size_t)16 * NRES * NRES * 2, stream);
  softmax_kernel<<<dim3(48, 12), 256, 0, stream>>>(qfeat, kfeat, qq2, kk2, a2dbuf, mask, attn);

  // 5) attn @ [v_s | v_pt]
  resv_kernel<<<dim3(12, 12), 128, 0, stream>>>(attn, vcat, finalbuf, resptg);

  // 6) attn @ inputs_2d: second streaming pass over inputs_2d
  res2d_kernel<<<NRES, 256, 0, stream>>>(attn, in2d, finalbuf);

  // 7) Inverse-frame points + norms
  ptlocal_kernel<<<36, 256, 0, stream>>>(resptg, quat, trans, finalbuf);

  // 8) Output projection: out = finalbuf @ wout + bout
  gemm_wmma_kernel<<<dim3(12, 6), 128, 0, stream>>>(finalbuf, wout, bout, out,
                                                    NRES, NOUT2, NFIN, NFIN, NOUT2, NOUT2, 0);
}
